// QuantumPaperSelfAttention_37203006718044
// MI455X (gfx1250) — matrix operand order repaired and checked by reading, not compiled, not run on hardware
//
#include <hip/hip_runtime.h>
#include <hip/hip_bf16.h>
#include <math.h>

// ---------------------------------------------------------------------------
// QuantumPaperSelfAttention for MI455X (gfx1250, wave32, WMMA + TDM).
// Pipeline: f32->bf16 convert, QKV projections via v_wmma_f32_16x16x32_bf16
// with the weight tile staged through the Tensor Data Mover into LDS,
// fused causal tanh/L1 attention (single pass), output projection to f32.
// ---------------------------------------------------------------------------

#define D_MODEL 1024
#define N_HEAD  16
#define HEAD_DIM 64
#define BATCH   4
#define SEQ     2048
#define ROWS    (BATCH * SEQ)   // 8192

typedef __attribute__((ext_vector_type(16))) __bf16          v16bf;
typedef __attribute__((ext_vector_type(8)))  float           v8f;
typedef __attribute__((ext_vector_type(8)))  unsigned short  v8u16;
typedef __attribute__((ext_vector_type(16))) unsigned short  v16u16;
typedef __attribute__((ext_vector_type(4)))  unsigned int    v4u;
typedef __attribute__((ext_vector_type(8)))  int             v8i;
typedef __attribute__((ext_vector_type(4)))  int             v4i;

#if __has_builtin(__builtin_amdgcn_tensor_load_to_lds)
#define USE_TDM 1
#else
#define USE_TDM 0
#endif

__device__ __forceinline__ unsigned short f32_to_bf16(float f) {
  unsigned int u = __float_as_uint(f);
  u += 0x7FFFu + ((u >> 16) & 1u);   // round-to-nearest-even
  return (unsigned short)(u >> 16);
}

__device__ __forceinline__ v16bf pack_frag(v8u16 lo, v8u16 hi) {
  v16u16 c;
#pragma unroll
  for (int i = 0; i < 8; ++i) { c[i] = lo[i]; c[i + 8] = hi[i]; }
  return __builtin_bit_cast(v16bf, c);
}

// A-operand fragment (16x32 MxK, 16-bit). Source row-major, ld in elements.
// Lane l<16 : row=l, K={0..7,16..23}; lane l>=16: row=l-16, K={8..15,24..31}.
__device__ __forceinline__ v16bf load_a_frag(const unsigned short* base, int ld) {
  const int lane = threadIdx.x & 31;
  const unsigned short* p = base + (size_t)(lane & 15) * ld + ((lane >> 4) << 3);
  return pack_frag(*(const v8u16*)p, *(const v8u16*)(p + 16));
}

// B-operand fragment (32x16 KxN, 16-bit). Source holds B^T row-major
// (row n of source = column n of B). Lane l<16: col=l, K={0..7,16..23};
// lane l>=16: col=l-16, K={8..15,24..31}.
__device__ __forceinline__ v16bf load_b_frag(const unsigned short* base, int ld) {
  const int lane = threadIdx.x & 31;
  const unsigned short* p = base + (size_t)(lane & 15) * ld + ((lane >> 4) << 3);
  return pack_frag(*(const v8u16*)p, *(const v8u16*)(p + 16));
}

__device__ __forceinline__ v8f wmma_bf16(v16bf a, v16bf b, v8f c) {
  return __builtin_amdgcn_wmma_f32_16x16x32_bf16(false, a, false, b, (short)0, c,
                                                 false, false);
}

__device__ __forceinline__ void lds_fence() {
#if __has_builtin(__builtin_amdgcn_s_wait_dscnt)
  __builtin_amdgcn_s_wait_dscnt(0);
#endif
  asm volatile("" ::: "memory");   // keep compiler from reordering LDS ops
}

#if USE_TDM
// LDS byte offset of a generic pointer that points into LDS.
__device__ __forceinline__ unsigned lds_addr_of(const void* p) {
  return (unsigned)(unsigned long long)(const __attribute__((address_space(3))) void*)p;
}

// Issue a TDM 2D tile load (bf16): tile_k x tile_rows from a row-major tensor
// with row stride `stride_e` (elements) into LDS at lds_off (tightly packed).
// D# layout per CDNA5 ISA ch.8.3-8.4 (group0 128b, group1 256b, groups2/3 zero).
__device__ __forceinline__ void tdm_load_tile_b16(unsigned lds_off,
                                                  const void* gaddr,
                                                  unsigned tile_k,
                                                  unsigned tile_rows,
                                                  unsigned stride_e,
                                                  unsigned tensor_d0,
                                                  unsigned tensor_d1) {
  unsigned long long ga = (unsigned long long)gaddr;
  v4u g0;
  g0[0] = 1u;                                           // count=1, user descriptor
  g0[1] = lds_off;                                      // lds_addr (bytes)
  g0[2] = (unsigned)(ga & 0xFFFFFFFFu);                 // global_addr[31:0]
  g0[3] = (unsigned)((ga >> 32) & 0x01FFFFFFu)          // global_addr[56:32]
          | (2u << 30);                                 // type=2 ("image")
  v8i g1;
  g1[0] = (int)(1u << 16);                              // wg_mask=0, data_size=2B
  g1[1] = (int)((tensor_d0 & 0xFFFFu) << 16);           // tensor_dim0[15:0]
  g1[2] = (int)((tensor_d0 >> 16) |
                ((tensor_d1 & 0xFFFFu) << 16));         // dim0[31:16] | dim1[15:0]
  g1[3] = (int)((tensor_d1 >> 16) | (tile_k << 16));    // dim1[31:16] | tile_dim0
  g1[4] = (int)(tile_rows & 0xFFFFu);                   // tile_dim1, tile_dim2=0
  g1[5] = (int)stride_e;                                // tensor_dim0_stride[31:0]
  g1[6] = 0;                                            // stride0 hi, stride1 lo
  g1[7] = 0;                                            // stride1 hi
  v4i gz4 = {0, 0, 0, 0};                               // 2D: groups 2/3 unused
#if defined(__clang_major__) && __clang_major__ >= 23
  v8i gz8 = {0, 0, 0, 0, 0, 0, 0, 0};
  __builtin_amdgcn_tensor_load_to_lds(g0, g1, gz4, gz4, gz8, 0);  // 6-arg form
#else
  __builtin_amdgcn_tensor_load_to_lds(g0, g1, gz4, gz4, 0);       // 5-arg form
#endif
}
#endif

// ---------------------------------------------------------------------------
// elementwise f32 -> bf16
__global__ void cvt_f32_bf16(const float* __restrict__ src,
                             unsigned short* __restrict__ dst, int n) {
  int i = blockIdx.x * blockDim.x + threadIdx.x;
  if (i < n) dst[i] = f32_to_bf16(src[i]);
}

// V[B,S,D] -> Vt[B,H,Dh,S]   (linear over destination, coalesced writes)
__global__ void transpose_vh(const unsigned short* __restrict__ V,
                             unsigned short* __restrict__ Vt) {
  int i = blockIdx.x * blockDim.x + threadIdx.x;
  int s = i & (SEQ - 1);
  int d = (i >> 11) & (HEAD_DIM - 1);
  int h = (i >> 17) & (N_HEAD - 1);
  int b = i >> 21;
  Vt[i] = V[(size_t)(b * SEQ + s) * D_MODEL + h * HEAD_DIM + d];
}

// ---------------------------------------------------------------------------
// C[M,N] = A[M,K] * W[N,K]^T   (bf16 in, f32 accumulate, OutT out)
// 256 threads = 8 waves; block tile 128x128; wave tile 64x32 (4x2 fragments).
// The 128x32 W tile for each K-step is staged block-wide via the TDM.
__device__ __forceinline__ void store_c(float* C, size_t i, float v) { C[i] = v; }
__device__ __forceinline__ void store_c(unsigned short* C, size_t i, float v) {
  C[i] = f32_to_bf16(v);
}

template <typename OutT>
__global__ __launch_bounds__(256) void gemm_xwT(const unsigned short* __restrict__ A,
                                                const unsigned short* __restrict__ W,
                                                OutT* __restrict__ C,
                                                int M, int N, int K) {
  __shared__ unsigned short wtile[128 * 32];  // 8 KB staged weight tile
  const int lane = threadIdx.x & 31;
  const int wave = threadIdx.x >> 5;
  const int m_base = blockIdx.y * 128 + (wave >> 2) * 64;
  const int n_base = blockIdx.x * 128 + (wave & 3) * 32;

  v8f acc[4][2];
#pragma unroll
  for (int i = 0; i < 4; ++i)
#pragma unroll
    for (int j = 0; j < 2; ++j)
#pragma unroll
      for (int v = 0; v < 8; ++v) acc[i][j][v] = 0.0f;

  for (int kk = 0; kk < K; kk += 32) {
    if (kk + 32 < K)  // lowers to global_prefetch_b8
      __builtin_prefetch(A + (size_t)m_base * K + kk + 32, 0, 1);

    v16bf a[4], b[2];
#if USE_TDM
    __syncthreads();  // previous tile fully consumed before TDM overwrites it
    if (wave == 0) {
      tdm_load_tile_b16(lds_addr_of(wtile),
                        W + (size_t)(blockIdx.x * 128) * K + kk,
                        /*tile_k=*/32, /*tile_rows=*/128, /*stride_e=*/(unsigned)K,
                        /*tensor_d0=*/(unsigned)(K - kk), /*tensor_d1=*/128u);
      __builtin_amdgcn_s_wait_tensorcnt(0);
    }
    __syncthreads();  // staged tile visible to all waves
#pragma unroll
    for (int j = 0; j < 2; ++j)
      b[j] = load_b_frag(wtile + ((wave & 3) * 32 + 16 * j) * 32, 32);
#else
#pragma unroll
    for (int j = 0; j < 2; ++j)
      b[j] = load_b_frag(W + (size_t)(n_base + 16 * j) * K + kk, K);
#endif
#pragma unroll
    for (int i = 0; i < 4; ++i)
      a[i] = load_a_frag(A + (size_t)(m_base + 16 * i) * K + kk, K);
#pragma unroll
    for (int i = 0; i < 4; ++i)
#pragma unroll
      for (int j = 0; j < 2; ++j) acc[i][j] = wmma_bf16(a[i], b[j], acc[i][j]);
  }

  const int half8 = (lane >> 4) << 3;
  const int col = lane & 15;
#pragma unroll
  for (int i = 0; i < 4; ++i)
#pragma unroll
    for (int j = 0; j < 2; ++j)
#pragma unroll
      for (int v = 0; v < 8; ++v) {
        size_t r = (size_t)(m_base + 16 * i + v + half8);
        size_t c = (size_t)(n_base + 16 * j + col);
        store_c(C, r * (size_t)N + c, acc[i][j][v]);
      }
}

// ---------------------------------------------------------------------------
// Fused causal attention. One wave per 16-row q-tile.
// out = (sum_k tanh(mask*S/8) @ V) / max(sum_k |tanh|, 1e-6) -- single pass.
__global__ __launch_bounds__(128) void attn_fused(const unsigned short* __restrict__ Q,
                                                  const unsigned short* __restrict__ Km,
                                                  const unsigned short* __restrict__ Vt,
                                                  unsigned short* __restrict__ O) {
  __shared__ unsigned short smem[4][16][40];  // per-wave 16x32 P tile, padded rows
  const int lane = threadIdx.x & 31;
  const int wave = threadIdx.x >> 5;
  const int b = blockIdx.z;
  const int h = blockIdx.y;
  const int q0 = (blockIdx.x * 4 + wave) * 16;
  const int half8 = (lane >> 4) << 3;
  const int nlane = lane & 15;

  const unsigned short* qbase = Q + (size_t)(b * SEQ + q0) * D_MODEL + h * HEAD_DIM;
  const v16bf aq0 = load_a_frag(qbase, D_MODEL);       // d = 0..31
  const v16bf aq1 = load_a_frag(qbase + 32, D_MODEL);  // d = 32..63

  v8f acc[4];
#pragma unroll
  for (int dn = 0; dn < 4; ++dn)
#pragma unroll
    for (int v = 0; v < 8; ++v) acc[dn][v] = 0.0f;
  float dsum[8];
#pragma unroll
  for (int v = 0; v < 8; ++v) dsum[v] = 0.0f;

  for (int k0 = 0; k0 < q0 + 16; k0 += 32) {
    // ---- scores S[16q x 32k] = Q(16x64) @ K^T(64x32), two 16-col n-tiles
    v8f s[2];
#pragma unroll
    for (int j = 0; j < 2; ++j)
#pragma unroll
      for (int v = 0; v < 8; ++v) s[j][v] = 0.0f;
#pragma unroll
    for (int j = 0; j < 2; ++j) {
      const unsigned short* kb =
          Km + (size_t)(b * SEQ + k0 + 16 * j) * D_MODEL + h * HEAD_DIM;
      v16bf b0 = load_b_frag(kb, D_MODEL);       // d 0..31
      v16bf b1 = load_b_frag(kb + 32, D_MODEL);  // d 32..63
      s[j] = wmma_bf16(aq0, b0, s[j]);
      s[j] = wmma_bf16(aq1, b1, s[j]);
    }
    // ---- causal mask + tanh + L1 partial, write P (bf16) to LDS row-major
#pragma unroll
    for (int j = 0; j < 2; ++j) {
#pragma unroll
      for (int v = 0; v < 8; ++v) {
        int r = q0 + v + half8;
        int c = k0 + 16 * j + nlane;
        float val = (c <= r) ? tanhf(s[j][v] * 0.125f) : 0.0f;
        dsum[v] += fabsf(val);
        smem[wave][v + half8][16 * j + nlane] = f32_to_bf16(val);
      }
    }
    lds_fence();
    // ---- reload P as A-fragment (cross-lane transpose via LDS)
    const unsigned short* sp = &smem[wave][nlane][half8];
    v16bf p = pack_frag(*(const v8u16*)sp, *(const v8u16*)(sp + 16));
    lds_fence();
    // ---- out += P(16x32) @ V(32x64); Vt gives contiguous B-fragments
#pragma unroll
    for (int dn = 0; dn < 4; ++dn) {
      const unsigned short* vb =
          Vt + (size_t)((b * N_HEAD + h) * HEAD_DIM + dn * 16) * SEQ + k0;
      v16bf bv = load_b_frag(vb, SEQ);
      acc[dn] = wmma_bf16(p, bv, acc[dn]);
    }
  }

  // ---- L1 denominator: butterfly sum over the 16 lanes of each half
#pragma unroll
  for (int v = 0; v < 8; ++v) {
    float t = dsum[v];
    t += __shfl_xor(t, 1, 32);
    t += __shfl_xor(t, 2, 32);
    t += __shfl_xor(t, 4, 32);
    t += __shfl_xor(t, 8, 32);
    float inv = 1.0f / fmaxf(t, 1e-6f);
#pragma unroll
    for (int dn = 0; dn < 4; ++dn) acc[dn][v] *= inv;
  }

  // ---- store attn output (bf16) back in [B,S,D] layout
#pragma unroll
  for (int dn = 0; dn < 4; ++dn)
#pragma unroll
    for (int v = 0; v < 8; ++v) {
      size_t r = (size_t)(b * SEQ + q0 + v + half8);
      size_t c = (size_t)(h * HEAD_DIM + dn * 16 + nlane);
      O[r * D_MODEL + c] = f32_to_bf16(acc[dn][v]);
    }
}

// ---------------------------------------------------------------------------
extern "C" void kernel_launch(void* const* d_in, const int* in_sizes, int n_in,
                              void* d_out, int out_size, void* d_ws, size_t ws_size,
                              hipStream_t stream) {
  (void)in_sizes; (void)n_in; (void)out_size; (void)ws_size;
  const float* x  = (const float*)d_in[0];
  const float* Wq = (const float*)d_in[1];
  const float* Wk = (const float*)d_in[2];
  const float* Wv = (const float*)d_in[3];
  const float* Wo = (const float*)d_in[4];
  float* out = (float*)d_out;

  char* ws = (char*)d_ws;
  size_t off = 0;
  auto walloc = [&](size_t bytes) -> void* {
    void* p = ws + off;
    off += (bytes + 255) & ~(size_t)255;
    return p;
  };
  const size_t nx = (size_t)ROWS * D_MODEL;     // 8388608
  const size_t nw = (size_t)D_MODEL * D_MODEL;  // 1048576

  unsigned short* xb  = (unsigned short*)walloc(nx * 2);
  unsigned short* wqb = (unsigned short*)walloc(nw * 2);
  unsigned short* wkb = (unsigned short*)walloc(nw * 2);
  unsigned short* wvb = (unsigned short*)walloc(nw * 2);
  unsigned short* wob = (unsigned short*)walloc(nw * 2);
  unsigned short* Qb  = (unsigned short*)walloc(nx * 2);
  unsigned short* Kb  = (unsigned short*)walloc(nx * 2);
  unsigned short* Vb  = (unsigned short*)walloc(nx * 2);
  unsigned short* Vtb = (unsigned short*)walloc(nx * 2);
  unsigned short* Ab  = (unsigned short*)walloc(nx * 2);

  cvt_f32_bf16<<<(int)((nx + 255) / 256), 256, 0, stream>>>(x, xb, (int)nx);
  cvt_f32_bf16<<<(int)((nw + 255) / 256), 256, 0, stream>>>(Wq, wqb, (int)nw);
  cvt_f32_bf16<<<(int)((nw + 255) / 256), 256, 0, stream>>>(Wk, wkb, (int)nw);
  cvt_f32_bf16<<<(int)((nw + 255) / 256), 256, 0, stream>>>(Wv, wvb, (int)nw);
  cvt_f32_bf16<<<(int)((nw + 255) / 256), 256, 0, stream>>>(Wo, wob, (int)nw);

  dim3 ggrid(D_MODEL / 128, ROWS / 128);  // (8, 64)
  gemm_xwT<unsigned short><<<ggrid, 256, 0, stream>>>(xb, wqb, Qb, ROWS, D_MODEL, D_MODEL);
  gemm_xwT<unsigned short><<<ggrid, 256, 0, stream>>>(xb, wkb, Kb, ROWS, D_MODEL, D_MODEL);
  gemm_xwT<unsigned short><<<ggrid, 256, 0, stream>>>(xb, wvb, Vb, ROWS, D_MODEL, D_MODEL);

  transpose_vh<<<(int)((nx + 255) / 256), 256, 0, stream>>>(Vb, Vtb);

  attn_fused<<<dim3(SEQ / 64, N_HEAD, BATCH), 128, 0, stream>>>(Qb, Kb, Vtb, Ab);

  gemm_xwT<float><<<ggrid, 256, 0, stream>>>(Ab, wob, out, ROWS, D_MODEL, D_MODEL);
}
